// MultiHeadAttention_34016140984782
// MI455X (gfx1250) — matrix operand order repaired and checked by reading, not compiled, not run on hardware
//
#include <hip/hip_runtime.h>
#include <hip/hip_bf16.h>
#include <cstdint>

// ---------- vector types ----------
typedef __attribute__((ext_vector_type(16))) __bf16          v16bf;
typedef __attribute__((ext_vector_type(8)))  float           v8f;
typedef __attribute__((ext_vector_type(4)))  unsigned short  u16x4;
typedef __attribute__((ext_vector_type(8)))  unsigned short  u16x8;
typedef __attribute__((ext_vector_type(16))) unsigned short  u16x16;
typedef __attribute__((ext_vector_type(4)))  unsigned int    u32x4;
typedef __attribute__((ext_vector_type(8)))  int             i32x8;
typedef __attribute__((ext_vector_type(4)))  int             i32x4;

__device__ __forceinline__ unsigned short f32_to_bf16(float f) {
  unsigned int u = __builtin_bit_cast(unsigned int, f);
  u += 0x7FFFu + ((u >> 16) & 1u);   // round-to-nearest-even
  return (unsigned short)(u >> 16);
}

// Truncate a generic pointer to an LDS byte offset (addr[31:0] per flat->LDS map)
__device__ __forceinline__ unsigned lds_off(const void* p) {
  return (unsigned)(uintptr_t)p;
}

// Async global->LDS 16B copy (per-lane addresses), tracked by ASYNCcnt.
__device__ __forceinline__ void async_copy_b128(unsigned lds, const void* g) {
  asm volatile("global_load_async_to_lds_b128 %0, %1, off"
               :: "v"(lds), "v"((unsigned long long)(uintptr_t)g)
               : "memory");
}
__device__ __forceinline__ void wait_asynccnt0() {
  asm volatile("s_wait_asynccnt 0" ::: "memory");
}

// Tensor Data Mover: 2D tile load (bf16 elements), wave-uniform arguments.
// D# packing per CDNA5 ISA 8.3/8.4: group0 {count=1, lds_addr, global_addr, type=2},
// group1 {data_size=2B, tensor_dim0/1, tile_dim0/1, tensor_dim0_stride}.
// This toolchain exposes the 6-arg builtin: (g0, g1, g2, g3, g4, cpol).
__device__ __forceinline__ void tdm_load_tile_2d_bf16(
    unsigned lds_addr, const void* gptr,
    unsigned tile_d0, unsigned tile_d1,
    unsigned tensor_d0, unsigned tensor_d1,
    unsigned long long d0_stride_elems)
{
  unsigned long long ga = (unsigned long long)(uintptr_t)gptr;
  u32x4 g0;
  g0[0] = 1u;                                                  // count=1, user D#
  g0[1] = lds_addr;                                            // lds_addr [63:32]
  g0[2] = (unsigned)(ga & 0xFFFFFFFFu);                        // global_addr lo
  g0[3] = (unsigned)((ga >> 32) & 0x01FFFFFFu) | (2u << 30);   // addr hi | type=2
  i32x8 g1;
  g1[0] = (int)(1u << 16);                                     // wg_mask=0, data_size=1 (2B)
  g1[1] = (int)(tensor_d0 << 16);                              // tensor_dim0[15:0] @63:48
  g1[2] = (int)((tensor_d0 >> 16) | (tensor_d1 << 16));        // dim0 hi | dim1 lo
  g1[3] = (int)((tensor_d1 >> 16) | (tile_d0 << 16));          // dim1 hi | tile_dim0
  g1[4] = (int)(tile_d1);                                      // tile_dim1 | tile_dim2=0
  g1[5] = (int)(unsigned)(d0_stride_elems & 0xFFFFFFFFu);      // dim0_stride lo32
  g1[6] = (int)(unsigned)((d0_stride_elems >> 32) & 0xFFFFu);  // dim0_stride hi16
  g1[7] = 0;                                                   // dim1_stride=0
  i32x4 z4 = {0, 0, 0, 0};
  i32x8 z8 = {0, 0, 0, 0, 0, 0, 0, 0};
  __builtin_amdgcn_tensor_load_to_lds(g0, g1, z4, z4, z8, 0);
}

// A fragment (16x32 bf16): lane&15 = row M, lane>>4 selects K {0..7,16..23} vs {8..15,24..31}
__device__ __forceinline__ v16bf lds_load_a16x32(const unsigned short* base, int stride, int lane) {
  const int row = lane & 15, half = lane >> 4;
  u16x8 lo = *(const u16x8*)(base + row * stride + half * 8);
  u16x8 hi = *(const u16x8*)(base + row * stride + 16 + half * 8);
  u16x16 c = __builtin_shufflevector(lo, hi, 0,1,2,3,4,5,6,7,8,9,10,11,12,13,14,15);
  return __builtin_bit_cast(v16bf, c);
}

// B fragment (32x16 bf16): LDS holds columns contiguous-in-K; lane&15 = column N,
// lane>>4 selects K {0..7,16..23} vs {8..15,24..31} -> two 16B LDS loads.
__device__ __forceinline__ v16bf lds_load_b32x16(const unsigned short* base, int stride, int lane) {
  const int col = lane & 15, half = lane >> 4;
  u16x8 lo = *(const u16x8*)(base + col * stride + half * 8);
  u16x8 hi = *(const u16x8*)(base + col * stride + 16 + half * 8);
  u16x16 c = __builtin_shufflevector(lo, hi, 0,1,2,3,4,5,6,7,8,9,10,11,12,13,14,15);
  return __builtin_bit_cast(v16bf, c);
}

#define WMMA_BF16(A, B, C) \
  __builtin_amdgcn_wmma_f32_16x16x32_bf16(false, (A), false, (B), (short)0, (C), false, false)

// ============================================================================
// GEMM: Out[M,N] = A[M,K] * B[K,N] + bias  (bf16 WMMA, fp32 accumulate)
// Block tile 128x128, BK=32, 256 threads = 8 waves (2 M x 4 N), wave tile 64x32.
// ============================================================================
template<bool A_BF16, bool OUT_MASKED_F32>
__global__ __launch_bounds__(256) void gemm_bias_kernel(
    const void* __restrict__ Ap, const float* __restrict__ Bp,
    const float* __restrict__ bias, const int* __restrict__ rmask,
    void* __restrict__ Outp, int M, int N, int K)
{
  __shared__ __align__(32) unsigned short As[128 * 32];   // [m][k]
  __shared__ __align__(32) unsigned short Bs[128 * 32];   // [n][k] (transposed)

  const int tid  = threadIdx.x;
  const int wave = tid >> 5, lane = tid & 31;
  const int wm = wave & 1, wn = wave >> 1;
  const int bn = blockIdx.x * 128;
  const int bm = blockIdx.y * 128;

  v8f acc[4][2] = {};

  for (int kt = 0; kt < K; kt += 32) {
    // prefetch next K-step tiles into cache (global_prefetch_b8)
    if (kt + 32 < K) {
      if constexpr (A_BF16)
        __builtin_prefetch(&((const unsigned short*)Ap)[(size_t)(bm + (tid >> 1)) * K + kt + 32], 0, 3);
      else
        __builtin_prefetch(&((const float*)Ap)[(size_t)(bm + (tid >> 1)) * K + kt + 32], 0, 3);
      __builtin_prefetch(&Bp[(size_t)(kt + 32 + (tid >> 5)) * N + bn + (tid & 31) * 4], 0, 3);
    }

    // ---- stage A tile 128x32 ----
    if constexpr (A_BF16) {
      // pure bf16 copy: async global->LDS, 512 x 16B chunks
      const unsigned short* A = (const unsigned short*)Ap;
      #pragma unroll
      for (int i = 0; i < 2; ++i) {
        int idx = tid + i * 256;
        int r = idx >> 2, c8 = idx & 3;          // 4 chunks of 8 bf16 per 32-elem row
        async_copy_b128(lds_off(&As[r * 32 + c8 * 8]),
                        &A[(size_t)(bm + r) * K + kt + c8 * 8]);
      }
    } else {
      const float* A = (const float*)Ap;
      #pragma unroll
      for (int i = 0; i < 4; ++i) {
        int idx = tid + i * 256;
        int r = idx >> 3, c4 = idx & 7;
        const float4 v = *(const float4*)&A[(size_t)(bm + r) * K + kt + c4 * 4];
        u16x4 h = { f32_to_bf16(v.x), f32_to_bf16(v.y), f32_to_bf16(v.z), f32_to_bf16(v.w) };
        *(u16x4*)&As[r * 32 + c4 * 4] = h;
      }
    }
    // ---- stage B tile 32x128, transposed to [n][k] with fp32->bf16 convert ----
    #pragma unroll
    for (int i = 0; i < 4; ++i) {
      int idx = tid + i * 256;
      int kr = idx >> 5, n4 = idx & 31;
      const float4 v = *(const float4*)&Bp[(size_t)(kt + kr) * N + bn + n4 * 4];
      Bs[(n4 * 4 + 0) * 32 + kr] = f32_to_bf16(v.x);
      Bs[(n4 * 4 + 1) * 32 + kr] = f32_to_bf16(v.y);
      Bs[(n4 * 4 + 2) * 32 + kr] = f32_to_bf16(v.z);
      Bs[(n4 * 4 + 3) * 32 + kr] = f32_to_bf16(v.w);
    }
    if constexpr (A_BF16) wait_asynccnt0();
    __syncthreads();

    v16bf a[4], b[2];
    #pragma unroll
    for (int mt = 0; mt < 4; ++mt)
      a[mt] = lds_load_a16x32(&As[(wm * 64 + mt * 16) * 32], 32, lane);
    #pragma unroll
    for (int nt = 0; nt < 2; ++nt)
      b[nt] = lds_load_b32x16(&Bs[(wn * 32 + nt * 16) * 32], 32, lane);
    #pragma unroll
    for (int mt = 0; mt < 4; ++mt)
      #pragma unroll
      for (int nt = 0; nt < 2; ++nt)
        acc[mt][nt] = WMMA_BF16(a[mt], b[nt], acc[mt][nt]);
    __syncthreads();
  }

  // ---- epilogue: C/D layout -> lanes 0-15: M=i, lanes 16-31: M=i+8, N=lane&15
  const int half = lane >> 4, nl = lane & 15;
  #pragma unroll
  for (int mt = 0; mt < 4; ++mt) {
    #pragma unroll
    for (int nt = 0; nt < 2; ++nt) {
      #pragma unroll
      for (int i = 0; i < 8; ++i) {
        int row = bm + wm * 64 + mt * 16 + half * 8 + i;
        int col = bn + wn * 32 + nt * 16 + nl;
        float v = acc[mt][nt][i] + bias[col];
        if constexpr (OUT_MASKED_F32) {
          v *= (float)rmask[row];
          ((float*)Outp)[(size_t)row * N + col] = v;
        } else {
          ((unsigned short*)Outp)[(size_t)row * N + col] = f32_to_bf16(v);
        }
      }
    }
  }
}

// ============================================================================
// Flash attention core. One block per (batch, head, 64-row query tile).
// Q tile staged by async global->LDS; K tile staged by the Tensor Data Mover;
// V^T staged with vector global loads + strided LDS stores (transpose).
// Scores through LDS fp32 with scale + v_mask + causal fused; online softmax;
// PV via WMMA.
// ============================================================================
__global__ __launch_bounds__(256) void attn_kernel(
    const unsigned short* __restrict__ Qw, const unsigned short* __restrict__ Kw,
    const unsigned short* __restrict__ Vw, const int* __restrict__ vmask,
    unsigned short* __restrict__ AO, int S)
{
  constexpr int H = 16, D = 64, HD = 1024;
  __shared__ __align__(32) unsigned short Qs [64 * 64];  // [q][d]
  __shared__ __align__(32) unsigned short Ks [64 * 64];  // [key][d]  (TDM target)
  __shared__ __align__(32) unsigned short Vts[64 * 64];  // [d][key]
  __shared__ __align__(32) unsigned short Ps [64 * 64];  // [q][key] probabilities bf16
  __shared__ float Ssc[64 * 64];                         // scores fp32; reused for O combine
  __shared__ float rowm[64], rowl[64], alphas[64], pen[64];

  const int tid  = threadIdx.x;
  const int wave = tid >> 5, lane = tid & 31;
  const int half = lane >> 4, nl = lane & 15;
  const int qb = wave & 3, kh = wave >> 2;

  const int QT = S / 64;
  const int b   = blockIdx.x / (H * QT);
  const int rem = blockIdx.x % (H * QT);
  const int h   = rem / QT;
  const int qt  = rem % QT;

  const size_t base = ((size_t)b * S) * HD + (size_t)h * D;

  // ---- load Q tile: async global->LDS (layout-preserving bf16 copy) ----
  #pragma unroll
  for (int i = 0; i < 2; ++i) {
    int idx = tid + i * 256;                 // 512 x 16B chunks
    int r = idx >> 3, d8 = idx & 7;          // 8 chunks of 8 bf16 per 64-elem row
    async_copy_b128(lds_off(&Qs[r * 64 + d8 * 8]),
                    &Qw[base + (size_t)(qt * 64 + r) * HD + d8 * 8]);
  }
  wait_asynccnt0();
  if (tid < 64) { rowm[tid] = -3.0e38f; rowl[tid] = 0.0f; }

  v8f oacc[4] = {};   // wave-local partial O: 16 q x 64 d over this key-half

  for (int kb = 0; kb <= qt; ++kb) {
    // ---- stage K tile [key][d] via Tensor Data Mover (wave 0, uniform D#) ----
    if (wave == 0) {
      tdm_load_tile_2d_bf16(lds_off(&Ks[0]),
                            &Kw[base + (size_t)(kb * 64) * HD],
                            /*tile*/ 64, 64, /*tensor*/ 64, 64,
                            /*dim0 stride*/ (unsigned long long)HD);
      __builtin_amdgcn_s_wait_tensorcnt(0);
    }
    // ---- stage V^T tile [d][key]: vector loads along d, strided LDS stores ----
    // 512 chunks of 8 bf16; thread's chunk = (key, d8). Loads are b128 and all
    // issue before the LDS stores consume them (no per-element load waits).
    {
      u16x8 vv[2];
      int key[2], d8[2];
      #pragma unroll
      for (int i = 0; i < 2; ++i) {
        int idx = tid + i * 256;
        key[i] = idx >> 3;                   // 0..63
        d8[i]  = idx & 7;                    // 8-elem group along d
        vv[i] = *(const u16x8*)&Vw[base + (size_t)(kb * 64 + key[i]) * HD + d8[i] * 8];
      }
      #pragma unroll
      for (int i = 0; i < 2; ++i)
        #pragma unroll
        for (int j = 0; j < 8; ++j)
          Vts[(d8[i] * 8 + j) * 64 + key[i]] = vv[i][j];
    }
    if (tid < 64)
      pen[tid] = (1.0f - (float)vmask[b * S + kb * 64 + tid]) * 1.0e12f;
    __syncthreads();

    // ---- S = Q K^T for this wave's 16q x 32k slab ----
    v8f s[2] = {};
    #pragma unroll
    for (int ks = 0; ks < 2; ++ks) {
      v16bf a = lds_load_a16x32(&Qs[(qb * 16) * 64 + ks * 32], 64, lane);
      #pragma unroll
      for (int nt = 0; nt < 2; ++nt) {
        v16bf bk = lds_load_b32x16(&Ks[(kh * 32 + nt * 16) * 64 + ks * 32], 64, lane);
        s[nt] = WMMA_BF16(a, bk, s[nt]);
      }
    }
    #pragma unroll
    for (int nt = 0; nt < 2; ++nt)
      #pragma unroll
      for (int i = 0; i < 8; ++i) {
        int ql = qb * 16 + half * 8 + i;
        int kl = kh * 32 + nt * 16 + nl;
        int gq = qt * 64 + ql, gk = kb * 64 + kl;
        Ssc[ql * 64 + kl] =
            s[nt][i] * 0.125f - pen[kl] - (gk > gq ? 1.0e12f : 0.0f);
      }
    __syncthreads();

    // ---- online softmax: one thread per query row ----
    if (tid < 64) {
      const int q = tid;
      float m = rowm[q];
      #pragma unroll 8
      for (int k2 = 0; k2 < 64; ++k2) m = fmaxf(m, Ssc[q * 64 + k2]);
      float alpha = __expf(rowm[q] - m);
      float l = rowl[q] * alpha;
      #pragma unroll 8
      for (int k2 = 0; k2 < 64; ++k2) {
        float p = __expf(Ssc[q * 64 + k2] - m);
        l += p;
        Ps[q * 64 + k2] = f32_to_bf16(p);
      }
      rowm[q] = m; rowl[q] = l; alphas[q] = alpha;
    }
    __syncthreads();

    // ---- rescale running O, then O += P V ----
    {
      const float* ap = &alphas[qb * 16 + half * 8];
      float av[8];
      #pragma unroll
      for (int i = 0; i < 8; ++i) av[i] = ap[i];
      #pragma unroll
      for (int nt = 0; nt < 4; ++nt)
        #pragma unroll
        for (int i = 0; i < 8; ++i) oacc[nt][i] *= av[i];

      v16bf a = lds_load_a16x32(&Ps[(qb * 16) * 64 + kh * 32], 64, lane);
      #pragma unroll
      for (int nt = 0; nt < 4; ++nt) {
        v16bf bv = lds_load_b32x16(&Vts[(nt * 16) * 64 + kh * 32], 64, lane);
        oacc[nt] = WMMA_BF16(a, bv, oacc[nt]);
      }
    }
    __syncthreads();
  }

  // ---- combine the two key-half partial sums through LDS, normalize, store ----
  if (kh == 0) {
    #pragma unroll
    for (int nt = 0; nt < 4; ++nt)
      #pragma unroll
      for (int i = 0; i < 8; ++i)
        Ssc[(qb * 16 + half * 8 + i) * 64 + nt * 16 + nl] = oacc[nt][i];
  }
  __syncthreads();
  if (kh == 1) {
    #pragma unroll
    for (int nt = 0; nt < 4; ++nt)
      #pragma unroll
      for (int i = 0; i < 8; ++i)
        Ssc[(qb * 16 + half * 8 + i) * 64 + nt * 16 + nl] += oacc[nt][i];
  }
  __syncthreads();

  #pragma unroll
  for (int i = 0; i < 16; ++i) {
    int idx = tid + i * 256;
    int q = idx >> 6, d = idx & 63;
    float v = Ssc[q * 64 + d] / rowl[q];
    AO[base + (size_t)(qt * 64 + q) * HD + d] = f32_to_bf16(v);
  }
}

// ============================================================================
extern "C" void kernel_launch(void* const* d_in, const int* in_sizes, int n_in,
                              void* d_out, int out_size, void* d_ws, size_t ws_size,
                              hipStream_t stream) {
  const float* q      = (const float*)d_in[0];
  const float* k      = (const float*)d_in[1];
  const float* v      = (const float*)d_in[2];
  const int*   q_mask = (const int*)  d_in[3];
  const int*   v_mask = (const int*)  d_in[4];
  const float* Wq     = (const float*)d_in[5];
  const float* bq     = (const float*)d_in[6];
  const float* Wk     = (const float*)d_in[7];
  const float* bk     = (const float*)d_in[8];
  const float* Wv     = (const float*)d_in[9];
  const float* bv     = (const float*)d_in[10];
  const float* Wo     = (const float*)d_in[11];
  const float* bo     = (const float*)d_in[12];

  const int Bb = 4, S = 2048, Dm = 1024;
  const int M = Bb * S;                      // 8192 flattened rows

  // workspace: Q/K/V projections + attention output, all bf16 [M, 1024]
  unsigned short* Qw = (unsigned short*)d_ws;
  unsigned short* Kw = Qw + (size_t)M * Dm;
  unsigned short* Vw = Kw + (size_t)M * Dm;
  unsigned short* AO = Vw + (size_t)M * Dm;

  dim3 grid(Dm / 128, M / 128);              // (8, 64)
  gemm_bias_kernel<false, false><<<grid, 256, 0, stream>>>(q, Wq, bq, nullptr, Qw, M, Dm, Dm);
  gemm_bias_kernel<false, false><<<grid, 256, 0, stream>>>(k, Wk, bk, nullptr, Kw, M, Dm, Dm);
  gemm_bias_kernel<false, false><<<grid, 256, 0, stream>>>(v, Wv, bv, nullptr, Vw, M, Dm, Dm);

  attn_kernel<<<Bb * 16 * (S / 64), 256, 0, stream>>>(Qw, Kw, Vw, v_mask, AO, S);

  gemm_bias_kernel<true, true><<<grid, 256, 0, stream>>>(AO, Wo, bo, q_mask, d_out, M, Dm, Dm);
}
